// PairwiseBilinear_31688268710472
// MI455X (gfx1250) — compile-verified, matrix operand order checked by reading, not run on hardware
//
#include <hip/hip_runtime.h>
#include <hip/hip_fp16.h>

typedef __attribute__((ext_vector_type(16))) _Float16 v16h;
typedef __attribute__((ext_vector_type(8)))  _Float16 v8h;
typedef __attribute__((ext_vector_type(8)))  float    v8f;

#define DU   257          // real K (with bias)
#define DP   288          // padded K (9 * 32)
#define OF   32           // out features
#define BB   8
#define LL   512
#define BL   (BB*LL)      // 4096 rows of u / v
#define N1   (OF*DP)      // 9216  : stage-1 N  (o, v-padded)
#define N2   (LL*OF)      // 16384 : stage-2 N per batch (l, o)

#define NGRP1 (N1/64)     // 144 stage-1 n-groups
#define NGRP2 (N2/64)     // 256 stage-2 n-groups per batch

#define SROW 296          // LDS row stride in halves (592 B): conflict-free banks

// ---- workspace layout (bytes) ----
#define OFF_UH 0u
#define SZ_UH  (BL*DP*2u)                 // 2,359,296
#define OFF_VH (OFF_UH + SZ_UH)
#define SZ_VH  SZ_UH
#define OFF_WT (OFF_VH + SZ_VH)
#define SZ_WT  (OF*DP*DP*2u)              // 5,308,416
#define OFF_I  (OFF_WT + SZ_WT)           // 10,027,008
// I: BL * N1 fp16 = 75,497,472 bytes ; total ws use ~85.5 MB

#define WMMA_F16(a, b, c) \
    __builtin_amdgcn_wmma_f32_16x16x32_f16(false, (a), false, (b), (short)0, (c), false, false)

// Build one WMMA A/B operand for this lane from two 16B chunks (+0, +16 halves).
static __device__ __forceinline__ v16h load_op(const _Float16* base) {
    v8h lo = *reinterpret_cast<const v8h*>(base);
    v8h hi = *reinterpret_cast<const v8h*>(base + 16);
    v16h r;
#pragma unroll
    for (int i = 0; i < 8; ++i) { r[i] = lo[i]; r[i + 8] = hi[i]; }
    return r;
}

// ---------------- packing kernels ----------------
__global__ void __launch_bounds__(256)
pack_uv(const float* __restrict__ x, _Float16* __restrict__ xh, int n) {
    int i = blockIdx.x * 256 + threadIdx.x;
    if (i >= n) return;
    int j = i % DP, r = i / DP;
    float val = (j < 256) ? x[r * 256 + j] : ((j == 256) ? 1.0f : 0.0f);
    xh[i] = (_Float16)val;
}

// Wt[o][v][u]  <-  W[u][o][v], zero padded to DPxDP
__global__ void __launch_bounds__(256)
pack_w(const float* __restrict__ w, _Float16* __restrict__ wt, int n) {
    int i = blockIdx.x * 256 + threadIdx.x;
    if (i >= n) return;
    int u = i % DP;
    int t = i / DP;
    int v = t % DP;
    int o = t / DP;
    float val = (u < DU && v < DU) ? w[((size_t)u * OF + o) * DU + v] : 0.0f;
    wt[i] = (_Float16)val;
}

// ---------------- stage 1: I = Uh (4096x288) * Wt^T  -> fp16 ----------------
// Block: 128 rows x 64 cols. A panel (128x288) + B panel (64x288) staged in LDS.
__global__ void __launch_bounds__(256)
stage1(const _Float16* __restrict__ A, const _Float16* __restrict__ Bm,
       _Float16* __restrict__ I) {
    __shared__ __align__(16) _Float16 sA[128 * SROW];  // 75,776 B
    __shared__ __align__(16) _Float16 sB[64 * SROW];   // 37,888 B

    const int tid = threadIdx.x;
    const int bm  = blockIdx.x / NGRP1;         // 0..31
    const int ng  = blockIdx.x % NGRP1;         // 0..143
    const int m0b = bm * 128;
    const int n0  = ng * 64;

    // cooperative stage: A rows (128 * 36 chunks = 18 iters), B cols (9 iters)
#pragma unroll 2
    for (int j = tid; j < 128 * 36; j += 256) {
        int r = j / 36, q = (j % 36) * 8;
        *reinterpret_cast<v8h*>(&sA[r * SROW + q]) =
            *reinterpret_cast<const v8h*>(A + (size_t)(m0b + r) * DP + q);
    }
#pragma unroll 2
    for (int j = tid; j < 64 * 36; j += 256) {
        int c = j / 36, q = (j % 36) * 8;
        *reinterpret_cast<v8h*>(&sB[c * SROW + q]) =
            *reinterpret_cast<const v8h*>(Bm + (size_t)(n0 + c) * DP + q);
    }
    __syncthreads();

    const int lane = tid & 31, wib = tid >> 5;
    const int lm   = lane & 15;
    const int klo  = (lane >> 4) * 8;
    const _Float16* ap = &sA[(wib * 16 + lm) * SROW + klo];
    const _Float16* bp = &sB[lm * SROW + klo];

    v8f c0 = {}, c1 = {}, c2 = {}, c3 = {};
#pragma unroll
    for (int k = 0; k < DP; k += 32) {
        v16h a  = load_op(ap + k);
        v16h b0 = load_op(bp + k);
        v16h b1 = load_op(bp + 16 * SROW + k);
        v16h b2 = load_op(bp + 32 * SROW + k);
        v16h b3 = load_op(bp + 48 * SROW + k);
        c0 = WMMA_F16(a, b0, c0);
        c1 = WMMA_F16(a, b1, c1);
        c2 = WMMA_F16(a, b2, c2);
        c3 = WMMA_F16(a, b3, c3);
    }

    const int m0   = m0b + wib * 16;
    const int row0 = m0 + (lane >> 4) * 8;
    _Float16* op = I + (size_t)row0 * N1 + n0 + lm;
#pragma unroll
    for (int i = 0; i < 8; ++i) {
        op[(size_t)i * N1 +  0] = (_Float16)c0[i];
        op[(size_t)i * N1 + 16] = (_Float16)c1[i];
        op[(size_t)i * N1 + 32] = (_Float16)c2[i];
        op[(size_t)i * N1 + 48] = (_Float16)c3[i];
    }
}

// ---------------- stage 2: scores[b][m][l*32+o] = Vh_b * I_b^T ----------------
// Same LDS-staged skeleton; B columns gathered from I (n=(l,o) -> contiguous v).
__global__ void __launch_bounds__(256)
stage2(const _Float16* __restrict__ Vh, const _Float16* __restrict__ I,
       float* __restrict__ out) {
    __shared__ __align__(16) _Float16 sA[128 * SROW];
    __shared__ __align__(16) _Float16 sB[64 * SROW];

    const int tid = threadIdx.x;
    const int b   = blockIdx.x / (4 * NGRP2);
    const int rem = blockIdx.x % (4 * NGRP2);
    const int bm  = rem / NGRP2;                // 0..3
    const int ng  = rem % NGRP2;                // 0..255
    const int m0b = bm * 128;
    const int n0  = ng * 64;

#pragma unroll 2
    for (int j = tid; j < 128 * 36; j += 256) {
        int r = j / 36, q = (j % 36) * 8;
        *reinterpret_cast<v8h*>(&sA[r * SROW + q]) =
            *reinterpret_cast<const v8h*>(Vh + (size_t)(b * LL + m0b + r) * DP + q);
    }
#pragma unroll 2
    for (int j = tid; j < 64 * 36; j += 256) {
        int c = j / 36, q = (j % 36) * 8;
        int n = n0 + c, l = n >> 5, o = n & 31;
        *reinterpret_cast<v8h*>(&sB[c * SROW + q]) =
            *reinterpret_cast<const v8h*>(I + (size_t)(b * LL + l) * N1 + (size_t)o * DP + q);
    }
    __syncthreads();

    const int lane = tid & 31, wib = tid >> 5;
    const int lm   = lane & 15;
    const int klo  = (lane >> 4) * 8;
    const _Float16* ap = &sA[(wib * 16 + lm) * SROW + klo];
    const _Float16* bp = &sB[lm * SROW + klo];

    v8f c0 = {}, c1 = {}, c2 = {}, c3 = {};
#pragma unroll
    for (int k = 0; k < DP; k += 32) {
        v16h a  = load_op(ap + k);
        v16h b0 = load_op(bp + k);
        v16h b1 = load_op(bp + 16 * SROW + k);
        v16h b2 = load_op(bp + 32 * SROW + k);
        v16h b3 = load_op(bp + 48 * SROW + k);
        c0 = WMMA_F16(a, b0, c0);
        c1 = WMMA_F16(a, b1, c1);
        c2 = WMMA_F16(a, b2, c2);
        c3 = WMMA_F16(a, b3, c3);
    }

    const int m0   = m0b + wib * 16;
    const int row0 = m0 + (lane >> 4) * 8;
    float* op = out + (size_t)(b * LL + row0) * N2 + n0 + lm;
#pragma unroll
    for (int i = 0; i < 8; ++i) {
        op[(size_t)i * N2 +  0] = c0[i];
        op[(size_t)i * N2 + 16] = c1[i];
        op[(size_t)i * N2 + 32] = c2[i];
        op[(size_t)i * N2 + 48] = c3[i];
    }
}

extern "C" void kernel_launch(void* const* d_in, const int* in_sizes, int n_in,
                              void* d_out, int out_size, void* d_ws, size_t ws_size,
                              hipStream_t stream) {
    const float* u = (const float*)d_in[0];
    const float* v = (const float*)d_in[1];
    const float* w = (const float*)d_in[2];
    float* out = (float*)d_out;
    char* ws = (char*)d_ws;

    _Float16* Uh = (_Float16*)(ws + OFF_UH);
    _Float16* Vh = (_Float16*)(ws + OFF_VH);
    _Float16* Wt = (_Float16*)(ws + OFF_WT);
    _Float16* Ih = (_Float16*)(ws + OFF_I);

    int nuv = BL * DP;                         // 1,179,648
    pack_uv<<<(nuv + 255) / 256, 256, 0, stream>>>(u, Uh, nuv);
    pack_uv<<<(nuv + 255) / 256, 256, 0, stream>>>(v, Vh, nuv);

    int nw = OF * DP * DP;                     // 2,654,208
    pack_w<<<(nw + 255) / 256, 256, 0, stream>>>(w, Wt, nw);

    // stage1: 32 m-super-blocks * 144 n-groups (128x64 per block)
    stage1<<<32 * NGRP1, 256, 0, stream>>>(Uh, Wt, Ih);

    // stage2: 8 batches * 4 m-super-blocks * 256 n-groups (128x64 per block)
    stage2<<<BB * 4 * NGRP2, 256, 0, stream>>>(Vh, Ih, out);
}
